// MultiHeadedAttention_74397423501870
// MI455X (gfx1250) — compile-verified, matrix operand order checked by reading, not run on hardware
//
#include <hip/hip_runtime.h>
#include <hip/hip_bf16.h>

// MHA forward for B=4, S=1024, U=1024, H=16, D=64 on gfx1250 (MI455X).
// All matmuls run on v_wmma_f32_16x16x32_f16 (wave32 WMMA).

typedef __attribute__((ext_vector_type(16))) _Float16 v16h;
typedef __attribute__((ext_vector_type(8)))  float    v8f;
typedef int v4i_ __attribute__((vector_size(16)));   // matches async-LDS builtin param

#define B_  4
#define S_  1024
#define U_  1024
#define H_  16
#define D_  64
#define M_  (B_ * S_)   // 4096 rows for the projection GEMMs

// ---- optional gfx1250 async global->LDS path (guarded; manual fallback) ----
#if defined(__has_builtin)
# if __has_builtin(__builtin_amdgcn_global_load_async_to_lds_b128) && \
     __has_builtin(__builtin_amdgcn_s_wait_asynccnt)
#  define HAVE_ASYNC_LDS 1
# endif
#endif
#ifndef HAVE_ASYNC_LDS
# define HAVE_ASYNC_LDS 0
#endif

#if HAVE_ASYNC_LDS
__device__ __forceinline__ void async_copy16B(const _Float16* g, _Float16* l) {
    __builtin_amdgcn_global_load_async_to_lds_b128(
        (v4i_*)(void*)g, (v4i_*)(void*)l, 0, 0);
}
#endif

// ---------------------------------------------------------------------------
// WMMA fragment helpers (16-bit A-layout per CDNA5 ISA 7.12.2):
//   lane 0-15  : row M = lane,    halves 0-7 -> K 0-7,   halves 8-15 -> K 16-23
//   lane 16-31 : row M = lane-16, halves 0-7 -> K 8-15,  halves 8-15 -> K 24-31
// B fragment uses the symmetric layout with lane = N and identical K packing.
// All LDS tiles are stored K-contiguous so these gathers vectorize to
// ds_load_b128 (halves 0-7 and 8-15 are each 16 contiguous, 16B-aligned bytes).
// ---------------------------------------------------------------------------
__device__ __forceinline__ int frag_k(int h, int lane) {
    const int hi = (lane >> 4) & 1;
    return (h < 8) ? (h + (hi ? 8 : 0)) : ((h - 8) + (hi ? 24 : 16));
}

// Element at p[(r0 + laneRow) * ld + k0 + frag_k]  (row-major [row][k])
__device__ __forceinline__ v16h load_frag_row(const _Float16* p, int r0, int k0, int ld) {
    const int lane = threadIdx.x & 31;
    const _Float16* row = p + (size_t)(r0 + (lane & 15)) * ld + k0;
    v16h f;
#pragma unroll
    for (int h = 0; h < 16; ++h) f[h] = row[frag_k(h, lane)];
    return f;
}

__device__ __forceinline__ v8f wmma_f16(v16h a, v16h b, v8f c) {
    return __builtin_amdgcn_wmma_f32_16x16x32_f16(false, a, false, b, (short)0, c, false, false);
}

// ---------------------------------------------------------------------------
// GEMM + bias:  C[M,N] = A[M,K] @ W[K,N] + bias[N]
// Workgroup tile 128x128, BK=64 (two WMMA K-steps per stage -> 16 WMMAs per
// barrier pair). 256 threads = 8 waves (4x2), wave tile 32x64.
// ---------------------------------------------------------------------------
template <typename TA, typename TC>
__global__ __launch_bounds__(256)
void gemm_bias_kernel(const TA* __restrict__ A, const float* __restrict__ W,
                      const float* __restrict__ bias, TC* __restrict__ C,
                      int M, int N, int K)
{
    __shared__ _Float16 sA[128 * 64];   // [m][k]   16 KB
    __shared__ _Float16 sB[128 * 64];   // [n][k]   16 KB (W tile transposed)

    const int tid  = threadIdx.x;
    const int wave = tid >> 5;
    const int lane = tid & 31;
    const int m0   = blockIdx.y * 128;
    const int n0   = blockIdx.x * 128;
    const int wm   = (wave >> 1) * 32;  // wave row offset inside tile
    const int wn   = (wave & 1) * 64;   // wave col offset inside tile

    v8f acc[2][4];
#pragma unroll
    for (int i = 0; i < 2; ++i)
#pragma unroll
        for (int j = 0; j < 4; ++j)
#pragma unroll
            for (int r = 0; r < 8; ++r) acc[i][j][r] = 0.0f;

    const int KB = K / 64;
    for (int kb = 0; kb < KB; ++kb) {
        // Stage A tile (128x64): 2 threads per row, 32 contiguous elems each.
        {
            const int row = tid >> 1;
            const int ch  = (tid & 1) * 32;
            const TA* src = A + (size_t)(m0 + row) * K + kb * 64 + ch;
#pragma unroll
            for (int j = 0; j < 32; ++j)
                sA[row * 64 + ch + j] = (_Float16)(float)src[j];
            if (kb + 1 < KB)
                __builtin_prefetch(src + 64, 0, 1);   // global_prefetch_b8
        }
        // Stage W tile (64x128) transposed into sB[n][k].
        {
            const int k  = tid >> 2;
            const int nc = (tid & 3) * 32;
            const float* src = W + (size_t)(kb * 64 + k) * N + n0 + nc;
#pragma unroll
            for (int j = 0; j < 32; ++j)
                sB[(nc + j) * 64 + k] = (_Float16)src[j];
            if (kb + 1 < KB)
                __builtin_prefetch(src + (size_t)64 * N, 0, 1);
        }
        __syncthreads();

#pragma unroll
        for (int kk = 0; kk < 2; ++kk) {
            v16h afrag[2];
            afrag[0] = load_frag_row(sA, wm,      kk * 32, 64);
            afrag[1] = load_frag_row(sA, wm + 16, kk * 32, 64);
#pragma unroll
            for (int nt = 0; nt < 4; ++nt) {
                const v16h bfrag = load_frag_row(sB, wn + nt * 16, kk * 32, 64);
#pragma unroll
                for (int mt = 0; mt < 2; ++mt)
                    acc[mt][nt] = wmma_f16(afrag[mt], bfrag, acc[mt][nt]);
            }
        }
        __syncthreads();
    }

    // Epilogue. C/D layout: lane l, vgpr r -> M = r + (l>>4)*8, N = l&15.
    const int lg = (lane >> 4) * 8;
    const int nc = lane & 15;
#pragma unroll
    for (int mt = 0; mt < 2; ++mt)
#pragma unroll
        for (int nt = 0; nt < 4; ++nt) {
            const int col = n0 + wn + nt * 16 + nc;
            const float bv = bias[col];
#pragma unroll
            for (int r = 0; r < 8; ++r) {
                const int row = m0 + wm + mt * 16 + r + lg;
                C[(size_t)row * N + col] = (TC)(acc[mt][nt][r] + bv);
            }
        }
}

// ---------------------------------------------------------------------------
// Flash attention: grid (S/64, B*H), 128 threads = 4 waves, wave = 16 q-rows.
// Causal mask matches reference: masked scores = -1e10 before softmax.
// K tile staged via gfx1250 async global->LDS (raw f16 copy) when available;
// V tile staged transposed ([dim][key]) so PV B-fragments load as ds_load_b128.
// ---------------------------------------------------------------------------
__global__ __launch_bounds__(128)
void attn_kernel(const _Float16* __restrict__ Q, const _Float16* __restrict__ K,
                 const _Float16* __restrict__ V, _Float16* __restrict__ Ctx)
{
    __shared__ _Float16 sK [32 * 64];      // [key][d]      (QK^T B-fragments)
    __shared__ _Float16 sVt[64 * 32];      // [d][key]      (PV  B-fragments)
    __shared__ _Float16 sP [4][16 * 32];   // per-wave P relayout scratch

    const int tid  = threadIdx.x;
    const int wave = tid >> 5;
    const int lane = tid & 31;
    const int b    = blockIdx.y >> 4;
    const int h    = blockIdx.y & 15;
    const int qB   = blockIdx.x * 64;      // block's query base
    const int qW   = qB + wave * 16;       // wave's query base
    const float scale = 0.125f;            // 1/sqrt(64)

    const _Float16* Qb = Q + (size_t)b * S_ * U_ + h * D_;

    // Q fragments for this wave's 16 rows, K-dim 64 = two WMMA K-steps.
    v16h qf[2];
    qf[0] = load_frag_row(Qb, qW, 0,  U_);
    qf[1] = load_frag_row(Qb, qW, 32, U_);

    float m_i[8], l_i[8];
    v8f o[4];
#pragma unroll
    for (int r = 0; r < 8; ++r) { m_i[r] = -1e30f; l_i[r] = 0.0f; }
#pragma unroll
    for (int d4 = 0; d4 < 4; ++d4)
#pragma unroll
        for (int r = 0; r < 8; ++r) o[d4][r] = 0.0f;

    const int lg = (lane >> 4) * 8;
    const int nc = lane & 15;
    const int nKB = blockIdx.x * 2 + 2;    // 32-key blocks covering keys <= qB+63

    for (int jb = 0; jb < nKB; ++jb) {
        const int j0 = jb * 32;
        const int keyId = tid >> 2;
        const int dc    = (tid & 3) * 16;
        const size_t off = ((size_t)b * S_ + j0 + keyId) * U_ + h * D_ + dc;

        // V: coalesced global read, transposed LDS write -> sVt[d][key].
        {
            const _Float16* vsrc = V + off;
#pragma unroll
            for (int j = 0; j < 16; ++j)
                sVt[(dc + j) * 32 + keyId] = vsrc[j];
        }
        // K: raw f16 tile copy. Async DMA to LDS when the toolchain has it.
#if HAVE_ASYNC_LDS
        {
            const _Float16* ksrc = K + off;
            _Float16* kdst = &sK[keyId * 64 + dc];
            async_copy16B(ksrc,     kdst);       // halves 0..7
            async_copy16B(ksrc + 8, kdst + 8);   // halves 8..15
            __builtin_amdgcn_s_wait_asynccnt(0);
        }
#else
        {
            const _Float16* ksrc = K + off;
#pragma unroll
            for (int j = 0; j < 16; ++j)
                sK[keyId * 64 + dc + j] = ksrc[j];
        }
#endif
        __syncthreads();

        // S = Q @ K^T : 16x32 scores as two 16x16 accumulators.
        v8f s[2];
#pragma unroll
        for (int nt = 0; nt < 2; ++nt) {
#pragma unroll
            for (int r = 0; r < 8; ++r) s[nt][r] = 0.0f;
#pragma unroll
            for (int kk = 0; kk < 2; ++kk) {
                const v16h bf = load_frag_row(sK, nt * 16, kk * 32, 64);
                s[nt] = wmma_f16(qf[kk], bf, s[nt]);
            }
        }

        // Scale, causal mask, online softmax.
        float alpha[8];
#pragma unroll
        for (int r = 0; r < 8; ++r) {
            const int qRow = qW + r + lg;
            float s0 = s[0][r] * scale;
            float s1 = s[1][r] * scale;
            if (j0 + nc      > qRow) s0 = -1e10f;
            if (j0 + 16 + nc > qRow) s1 = -1e10f;
            float t = fmaxf(s0, s1);
            t = fmaxf(t, __shfl_xor(t, 1, 16));
            t = fmaxf(t, __shfl_xor(t, 2, 16));
            t = fmaxf(t, __shfl_xor(t, 4, 16));
            t = fmaxf(t, __shfl_xor(t, 8, 16));
            const float mn = fmaxf(m_i[r], t);
            alpha[r] = __expf(m_i[r] - mn);
            const float p0 = __expf(s0 - mn);
            const float p1 = __expf(s1 - mn);
            s[0][r] = p0;
            s[1][r] = p1;
            float rs = p0 + p1;
            rs += __shfl_xor(rs, 1, 16);
            rs += __shfl_xor(rs, 2, 16);
            rs += __shfl_xor(rs, 4, 16);
            rs += __shfl_xor(rs, 8, 16);
            l_i[r] = l_i[r] * alpha[r] + rs;
            m_i[r] = mn;
        }

        // Re-layout P (C-layout -> A-fragment) via per-wave LDS scratch.
        _Float16* Pw = sP[wave];
#pragma unroll
        for (int nt = 0; nt < 2; ++nt)
#pragma unroll
            for (int r = 0; r < 8; ++r)
                Pw[(r + lg) * 32 + nt * 16 + nc] = (_Float16)s[nt][r];
        const v16h pf = load_frag_row(Pw, 0, 0, 32);  // same-wave LDS is in-order

        // O = diag(alpha)*O + P @ V   (V fragment from transposed tile)
#pragma unroll
        for (int d4 = 0; d4 < 4; ++d4) {
#pragma unroll
            for (int r = 0; r < 8; ++r) o[d4][r] *= alpha[r];
            const v16h vf = load_frag_row(sVt, d4 * 16, 0, 32);
            o[d4] = wmma_f16(pf, vf, o[d4]);
        }
        __syncthreads();
    }

    // Normalize and write context (f16) back to workspace.
    _Float16* Cb = Ctx + (size_t)b * S_ * U_ + h * D_;
#pragma unroll
    for (int d4 = 0; d4 < 4; ++d4)
#pragma unroll
        for (int r = 0; r < 8; ++r) {
            const int qRow = qW + r + lg;
            Cb[(size_t)qRow * U_ + d4 * 16 + nc] = (_Float16)(o[d4][r] / l_i[r]);
        }
}

// ---------------------------------------------------------------------------
// Launcher
// ---------------------------------------------------------------------------
extern "C" void kernel_launch(void* const* d_in, const int* in_sizes, int n_in,
                              void* d_out, int out_size, void* d_ws, size_t ws_size,
                              hipStream_t stream)
{
    (void)in_sizes; (void)n_in; (void)out_size; (void)ws_size;

    const float* query = (const float*)d_in[0];
    const float* key   = (const float*)d_in[1];
    const float* value = (const float*)d_in[2];
    // d_in[3] = mask (causal tril) -- applied analytically in attn_kernel.
    const float* Wq = (const float*)d_in[4];
    const float* bq = (const float*)d_in[5];
    const float* Wk = (const float*)d_in[6];
    const float* bk = (const float*)d_in[7];
    const float* Wv = (const float*)d_in[8];
    const float* bv = (const float*)d_in[9];
    const float* Wo = (const float*)d_in[10];
    const float* bo = (const float*)d_in[11];
    float* out = (float*)d_out;

    const size_t nElem = (size_t)M_ * U_;           // 4M elements per tensor
    _Float16* Qh  = (_Float16*)d_ws;
    _Float16* Kh  = Qh + nElem;
    _Float16* Vh  = Kh + nElem;
    _Float16* Ctx = Vh + nElem;

    const dim3 gGemm(U_ / 128, M_ / 128);           // (8, 32)
    const dim3 bGemm(256);

    // Q/K/V projections: fp32 in, f16 out (workspace).
    gemm_bias_kernel<float, _Float16><<<gGemm, bGemm, 0, stream>>>(query, Wq, bq, Qh, M_, U_, U_);
    gemm_bias_kernel<float, _Float16><<<gGemm, bGemm, 0, stream>>>(key,   Wk, bk, Kh, M_, U_, U_);
    gemm_bias_kernel<float, _Float16><<<gGemm, bGemm, 0, stream>>>(value, Wv, bv, Vh, M_, U_, U_);

    // Causal flash attention per (b, h).
    const dim3 gAttn(S_ / 64, B_ * H_);             // (16, 64)
    attn_kernel<<<gAttn, dim3(128), 0, stream>>>(Qh, Kh, Vh, Ctx);

    // Output projection: f16 context in, fp32 out.
    gemm_bias_kernel<_Float16, float><<<gGemm, bGemm, 0, stream>>>(Ctx, Wo, bo, out, M_, U_, U_);
}
